// DeepSeekMoE_8589934592343
// MI455X (gfx1250) — matrix operand order repaired and checked by reading, not compiled, not run on hardware
//
#include <hip/hip_runtime.h>
#include <hip/hip_bf16.h>

#define DD 1024
#define EE 8
#define KACT 2
#define HH 4096
#define TT 4096
#define BM 128
#define BN 64
#define BK 32
#define MBMAX (TT / BM)   // 32 worst-case M-tiles per expert

typedef __attribute__((ext_vector_type(16))) __bf16 v16bf;
typedef __attribute__((ext_vector_type(8)))  float  v8f;

union FragAB { uint4 q[2]; v16bf v; };
union FragC  { float f[8];  v8f   v; };

__device__ __forceinline__ unsigned f2bf(float f) {
    union { __hip_bfloat16 b; unsigned short u; } cv;
    cv.b = __float2bfloat16(f);      // round-to-nearest-even
    return (unsigned)cv.u;
}

// Branch-free GELU: Abramowitz-Stegun 7.1.26 erf (|err| < 1.5e-7, far below
// the ~4e-3 relative noise already introduced by bf16 h storage).
__device__ __forceinline__ float gelu_exact(float v) {
    float z  = fabsf(v) * 0.70710678118654752f;
    float t  = __builtin_amdgcn_rcpf(fmaf(0.3275911f, z, 1.0f));
    float p  = fmaf(t, 1.061405429f, -1.453152027f);
    p = fmaf(t, p, 1.421413741f);
    p = fmaf(t, p, -0.284496736f);
    p = fmaf(t, p, 0.254829592f);
    p = p * t;
    float e  = __expf(-z * z);
    float er = fmaf(-p, e, 1.0f);          // erf(z), z >= 0
    er = copysignf(er, v);                 // erf is odd
    return 0.5f * v * (1.0f + er);
}

// ---------------------------------------------------------------------------
// Routing: one wave per token. 8 gate dots, softmax, top-2 (lowest-index tie
// break like lax.top_k), renormalize, atomic slot assignment per expert.
// ---------------------------------------------------------------------------
__global__ __launch_bounds__(256) void route_kernel(
    const float* __restrict__ x, const float* __restrict__ gw,
    const float* __restrict__ gb, const float* __restrict__ rbias,
    int* __restrict__ counts, int* __restrict__ etok,
    int* __restrict__ tslot, float* __restrict__ twt)
{
    const int lane = threadIdx.x & 31;
    const int wv   = threadIdx.x >> 5;
    const int t    = blockIdx.x * 8 + wv;

    float acc[EE];
    #pragma unroll
    for (int j = 0; j < EE; ++j) acc[j] = 0.0f;

    for (int d = lane; d < DD; d += 32) {
        float xv = x[(size_t)t * DD + d];
        const float4* g = (const float4*)(gw + (size_t)d * EE);
        float4 g0 = g[0], g1 = g[1];
        acc[0] += xv * g0.x; acc[1] += xv * g0.y;
        acc[2] += xv * g0.z; acc[3] += xv * g0.w;
        acc[4] += xv * g1.x; acc[5] += xv * g1.y;
        acc[6] += xv * g1.z; acc[7] += xv * g1.w;
    }
    #pragma unroll
    for (int off = 16; off >= 1; off >>= 1)
        #pragma unroll
        for (int j = 0; j < EE; ++j)
            acc[j] += __shfl_xor(acc[j], off, 32);

    if (lane == 0) {
        float lg[EE];
        #pragma unroll
        for (int j = 0; j < EE; ++j) lg[j] = acc[j] + gb[j] + rbias[j];
        float mx = lg[0];
        #pragma unroll
        for (int j = 1; j < EE; ++j) mx = fmaxf(mx, lg[j]);
        float ex[EE];
        #pragma unroll
        for (int j = 0; j < EE; ++j) ex[j] = __expf(lg[j] - mx);
        int i0 = 0;
        #pragma unroll
        for (int j = 1; j < EE; ++j) if (ex[j] > ex[i0]) i0 = j;
        int i1 = (i0 == 0) ? 1 : 0;
        #pragma unroll
        for (int j = 0; j < EE; ++j)
            if (j != i0 && ex[j] > ex[i1]) i1 = j;
        float inv = 1.0f / (ex[i0] + ex[i1]);
        float w0 = ex[i0] * inv, w1 = ex[i1] * inv;

        int s0 = atomicAdd(&counts[i0], 1);
        int s1 = atomicAdd(&counts[i1], 1);
        etok[i0 * TT + s0] = t;
        etok[i1 * TT + s1] = t;
        tslot[t * 2 + 0] = (i0 << 16) | s0;
        tslot[t * 2 + 1] = (i1 << 16) | s1;
        twt[t * 2 + 0] = w0;
        twt[t * 2 + 1] = w1;
    }
}

__global__ void scan_kernel(const int* __restrict__ counts, int* __restrict__ basep) {
    if (threadIdx.x == 0) {
        int s = 0;
        for (int e = 0; e < EE; ++e) { basep[e] = s; s += counts[e]; }
        basep[EE] = s;
    }
}

// ---------------------------------------------------------------------------
// GEMM1: H = gelu(Xg @ W1 + b1), bf16 output. A rows gathered per expert list
// (ROUTED) or identity (shared). fp32 -> bf16 conversion while staging LDS.
// ---------------------------------------------------------------------------
template <bool ROUTED>
__global__ __launch_bounds__(256) void ffn1_kernel(
    const float* __restrict__ X, const float* __restrict__ W,
    const float* __restrict__ B1, const int* __restrict__ counts,
    const int* __restrict__ basep, const int* __restrict__ etok,
    unsigned short* __restrict__ Hout)
{
    int e = 0, mt = blockIdx.y, cnt = TT;
    if (ROUTED) {
        e   = blockIdx.y / MBMAX;
        mt  = blockIdx.y % MBMAX;
        cnt = counts[e];
        if (mt * BM >= cnt) return;
    }
    const float* Wp = ROUTED ? W + (size_t)e * DD * HH : W;
    const float* Bp = ROUTED ? B1 + (size_t)e * HH : B1;
    const int n0 = blockIdx.x * BN;
    const int mBase = mt * BM;
    const int hRowBase = ROUTED ? basep[e] + mBase : mBase;

    __shared__ __align__(16) unsigned short As[BM][BK + 8];
    __shared__ __align__(16) unsigned short Bs[BN][BK + 8];

    const int tid  = threadIdx.x;
    const int lane = tid & 31;
    const int wid  = tid >> 5;
    const int wm   = (wid >> 1) * 32;   // wave M offset: 0..96
    const int wn   = (wid & 1) * 32;    // wave N offset: 0 or 32
    const int lm   = lane & 15;
    const int aKb  = (lane < 16) ? 0 : 8;   // A: K split {0..7,16..23}/{8..15,24..31}
    const int bKb  = (lane < 16) ? 0 : 8;   // B: K split {0..7,16..23}/{8..15,24..31}

    int tokc[BM / 32];
    #pragma unroll
    for (int rb = 0; rb < BM / 32; ++rb) {
        int gi = mBase + rb * 32 + (tid >> 3);
        tokc[rb] = ROUTED ? ((gi < cnt) ? etok[e * TT + gi] : 0) : gi;
    }

    // B staging decomposition: thread covers one column n, 8 consecutive k.
    const int bn = tid & 63;
    const int bk = (tid >> 6) * 8;

    FragC acc[2][2];
    #pragma unroll
    for (int i = 0; i < 2; ++i)
        #pragma unroll
        for (int j = 0; j < 2; ++j)
            #pragma unroll
            for (int k = 0; k < 8; ++k) acc[i][j].f[k] = 0.0f;

    for (int k0 = 0; k0 < DD; k0 += BK) {
        #pragma unroll
        for (int rb = 0; rb < BM / 32; ++rb) {            // A: 128x32 fp32->bf16
            int r = rb * 32 + (tid >> 3);
            int c = (tid & 7) << 2;
            float4 xv = *(const float4*)(X + (size_t)tokc[rb] * DD + k0 + c);
            uint2 pk;
            pk.x = f2bf(xv.x) | (f2bf(xv.y) << 16);
            pk.y = f2bf(xv.z) | (f2bf(xv.w) << 16);
            *(uint2*)&As[r][c] = pk;
        }
        {   // B: 32x64 fp32->bf16, coalesced b32 loads, one b128 LDS store
            const float* src = Wp + (size_t)(k0 + bk) * HH + n0 + bn;
            uint4 pk;
            pk.x = f2bf(src[0])              | (f2bf(src[(size_t)HH])     << 16);
            pk.y = f2bf(src[(size_t)2 * HH]) | (f2bf(src[(size_t)3 * HH]) << 16);
            pk.z = f2bf(src[(size_t)4 * HH]) | (f2bf(src[(size_t)5 * HH]) << 16);
            pk.w = f2bf(src[(size_t)6 * HH]) | (f2bf(src[(size_t)7 * HH]) << 16);
            *(uint4*)&Bs[bn][bk] = pk;
        }
        __syncthreads();

        FragAB a[2], b[2];
        #pragma unroll
        for (int i = 0; i < 2; ++i) {
            const unsigned short* ap = &As[wm + i * 16 + lm][0];
            a[i].q[0] = *(const uint4*)(ap + aKb);
            a[i].q[1] = *(const uint4*)(ap + 16 + aKb);
        }
        #pragma unroll
        for (int j = 0; j < 2; ++j) {
            const unsigned short* bp = &Bs[wn + j * 16 + lm][0];
            b[j].q[0] = *(const uint4*)(bp + bKb);
            b[j].q[1] = *(const uint4*)(bp + 16 + bKb);
        }
        #pragma unroll
        for (int i = 0; i < 2; ++i)
            #pragma unroll
            for (int j = 0; j < 2; ++j)
                acc[i][j].v = __builtin_amdgcn_wmma_f32_16x16x32_bf16(
                    false, a[i].v, false, b[j].v, (short)0, acc[i][j].v, false, false);
        __syncthreads();
    }

    #pragma unroll
    for (int i = 0; i < 2; ++i)
        #pragma unroll
        for (int j = 0; j < 2; ++j) {
            int n = n0 + wn + j * 16 + lm;
            float bias = Bp[n];
            #pragma unroll
            for (int kr = 0; kr < 8; ++kr) {
                int mIdx = wm + i * 16 + ((lane < 16) ? 0 : 8) + kr;
                if (!ROUTED || (mBase + mIdx) < cnt) {
                    float v = acc[i][j].f[kr] + bias;
                    Hout[(size_t)(hRowBase + mIdx) * HH + n] =
                        (unsigned short)f2bf(gelu_exact(v));
                }
            }
        }
}

// ---------------------------------------------------------------------------
// GEMM2: Y = H @ W2 + b2. A is bf16 (flat slot rows for routed, identity for
// shared) copied global->LDS with GLOBAL_LOAD_ASYNC_TO_LDS_B128 (ASYNCcnt).
// Shared writes `out` directly; routed writes ybuf.
// ---------------------------------------------------------------------------
template <bool ROUTED>
__global__ __launch_bounds__(256) void ffn2_kernel(
    const unsigned short* __restrict__ Hin, const float* __restrict__ W,
    const float* __restrict__ B2, const int* __restrict__ counts,
    const int* __restrict__ basep, float* __restrict__ Yout)
{
    int e = 0, mt = blockIdx.y, cnt = TT;
    if (ROUTED) {
        e   = blockIdx.y / MBMAX;
        mt  = blockIdx.y % MBMAX;
        cnt = counts[e];
        if (mt * BM >= cnt) return;
    }
    const float* Wp = ROUTED ? W + (size_t)e * HH * DD : W;
    const float* Bp = ROUTED ? B2 + (size_t)e * DD : B2;
    const int n0 = blockIdx.x * BN;
    const int mBase = mt * BM;
    const size_t rowBase = ROUTED ? (size_t)basep[e] + mBase : (size_t)mBase;

    __shared__ __align__(16) unsigned short As[BM][BK + 8];
    __shared__ __align__(16) unsigned short Bs[BN][BK + 8];

    const int tid  = threadIdx.x;
    const int lane = tid & 31;
    const int wid  = tid >> 5;
    const int wm   = (wid >> 1) * 32;
    const int wn   = (wid & 1) * 32;
    const int lm   = lane & 15;
    const int aKb  = (lane < 16) ? 0 : 8;
    const int bKb  = (lane < 16) ? 0 : 8;

    // Async A staging: thread owns row (tid>>1), 32-byte chunk (tid&1).
    size_t arow = rowBase + (tid >> 1);
    if (ROUTED && arow >= (size_t)TT * KACT) arow = (size_t)TT * KACT - 1;  // clamp tail
    const unsigned short* aRowPtr = Hin + arow * HH + ((tid & 1) * 16);
    const unsigned ldsA = (unsigned)(uintptr_t)&As[tid >> 1][(tid & 1) * 16];

    const int bn = tid & 63;
    const int bk = (tid >> 6) * 8;

    FragC acc[2][2];
    #pragma unroll
    for (int i = 0; i < 2; ++i)
        #pragma unroll
        for (int j = 0; j < 2; ++j)
            #pragma unroll
            for (int k = 0; k < 8; ++k) acc[i][j].f[k] = 0.0f;

    for (int k0 = 0; k0 < HH; k0 += BK) {
        {   // A: async global->LDS bf16 copy (2 x B128 per thread)
            unsigned long long ga = (unsigned long long)(uintptr_t)(aRowPtr + k0);
            asm volatile("global_load_async_to_lds_b128 %0, %1, off"
                         :: "v"(ldsA), "v"(ga) : "memory");
            asm volatile("global_load_async_to_lds_b128 %0, %1, off offset:16"
                         :: "v"(ldsA), "v"(ga) : "memory");
        }
        {   // B: 32x64 fp32->bf16
            const float* src = Wp + (size_t)(k0 + bk) * DD + n0 + bn;
            uint4 pk;
            pk.x = f2bf(src[0])              | (f2bf(src[(size_t)DD])     << 16);
            pk.y = f2bf(src[(size_t)2 * DD]) | (f2bf(src[(size_t)3 * DD]) << 16);
            pk.z = f2bf(src[(size_t)4 * DD]) | (f2bf(src[(size_t)5 * DD]) << 16);
            pk.w = f2bf(src[(size_t)6 * DD]) | (f2bf(src[(size_t)7 * DD]) << 16);
            *(uint4*)&Bs[bn][bk] = pk;
        }
        asm volatile("s_wait_asynccnt 0" ::: "memory");
        __syncthreads();

        FragAB a[2], b[2];
        #pragma unroll
        for (int i = 0; i < 2; ++i) {
            const unsigned short* ap = &As[wm + i * 16 + lm][0];
            a[i].q[0] = *(const uint4*)(ap + aKb);
            a[i].q[1] = *(const uint4*)(ap + 16 + aKb);
        }
        #pragma unroll
        for (int j = 0; j < 2; ++j) {
            const unsigned short* bp = &Bs[wn + j * 16 + lm][0];
            b[j].q[0] = *(const uint4*)(bp + bKb);
            b[j].q[1] = *(const uint4*)(bp + 16 + bKb);
        }
        #pragma unroll
        for (int i = 0; i < 2; ++i)
            #pragma unroll
            for (int j = 0; j < 2; ++j)
                acc[i][j].v = __builtin_amdgcn_wmma_f32_16x16x32_bf16(
                    false, a[i].v, false, b[j].v, (short)0, acc[i][j].v, false, false);
        __syncthreads();
    }

    #pragma unroll
    for (int i = 0; i < 2; ++i)
        #pragma unroll
        for (int j = 0; j < 2; ++j) {
            int n = n0 + wn + j * 16 + lm;
            float bias = Bp[n];
            #pragma unroll
            for (int kr = 0; kr < 8; ++kr) {
                int mIdx = wm + i * 16 + ((lane < 16) ? 0 : 8) + kr;
                if (!ROUTED || (mBase + mIdx) < cnt) {
                    Yout[(rowBase + mIdx) * DD + n] = acc[i][j].f[kr] + bias;
                }
            }
        }
}

// ---------------------------------------------------------------------------
// Combine: out[t] += w0 * ybuf[slot0] + w1 * ybuf[slot1]  (out already holds
// the shared-expert result incl. shared_b2; exp_b2 is already inside ybuf).
// ---------------------------------------------------------------------------
__global__ __launch_bounds__(256) void combine_kernel(
    const int* __restrict__ tslot, const float* __restrict__ twt,
    const int* __restrict__ basep, const float* __restrict__ ybuf,
    float* __restrict__ out)
{
    const int t = blockIdx.x;
    int id0 = tslot[t * 2 + 0], id1 = tslot[t * 2 + 1];
    float w0 = twt[t * 2 + 0], w1 = twt[t * 2 + 1];
    const float* y0 = ybuf + (size_t)(basep[id0 >> 16] + (id0 & 0xFFFF)) * DD;
    const float* y1 = ybuf + (size_t)(basep[id1 >> 16] + (id1 & 0xFFFF)) * DD;
    float* o = out + (size_t)t * DD;
    for (int d = threadIdx.x; d < DD; d += blockDim.x)
        o[d] += w0 * y0[d] + w1 * y1[d];
}

extern "C" void kernel_launch(void* const* d_in, const int* in_sizes, int n_in,
                              void* d_out, int out_size, void* d_ws, size_t ws_size,
                              hipStream_t stream) {
    const float* x          = (const float*)d_in[0];
    const float* gate_w     = (const float*)d_in[1];
    const float* gate_b     = (const float*)d_in[2];
    const float* route_bias = (const float*)d_in[3];
    const float* shared_w1  = (const float*)d_in[4];
    const float* shared_b1  = (const float*)d_in[5];
    const float* shared_w2  = (const float*)d_in[6];
    const float* shared_b2  = (const float*)d_in[7];
    const float* exp_w1     = (const float*)d_in[8];
    const float* exp_b1     = (const float*)d_in[9];
    const float* exp_w2     = (const float*)d_in[10];
    const float* exp_b2     = (const float*)d_in[11];
    float* out = (float*)d_out;

    // Workspace carve-up (~134.5 MB total)
    char* w = (char*)d_ws;
    int*   counts = (int*)w;            w += 256;
    int*   basep  = (int*)w;            w += 256;
    int*   etok   = (int*)w;            w += (size_t)EE * TT * sizeof(int);
    int*   tslot  = (int*)w;            w += (size_t)TT * KACT * sizeof(int);
    float* twt    = (float*)w;          w += (size_t)TT * KACT * sizeof(float);
    unsigned short* hsh = (unsigned short*)w; w += (size_t)TT * HH * sizeof(unsigned short);
    unsigned short* hex = (unsigned short*)w; w += (size_t)TT * KACT * HH * sizeof(unsigned short);
    float* ybuf   = (float*)w;          w += (size_t)TT * KACT * DD * sizeof(float);

    hipMemsetAsync(counts, 0, 256, stream);
    route_kernel<<<TT / 8, 256, 0, stream>>>(x, gate_w, gate_b, route_bias,
                                             counts, etok, tslot, twt);
    scan_kernel<<<1, 32, 0, stream>>>(counts, basep);

    ffn1_kernel<false><<<dim3(HH / BN, TT / BM), 256, 0, stream>>>(
        x, shared_w1, shared_b1, nullptr, nullptr, nullptr, hsh);
    ffn1_kernel<true><<<dim3(HH / BN, EE * MBMAX), 256, 0, stream>>>(
        x, exp_w1, exp_b1, counts, basep, etok, hex);

    ffn2_kernel<false><<<dim3(DD / BN, TT / BM), 256, 0, stream>>>(
        hsh, shared_w2, shared_b2, nullptr, nullptr, out);
    ffn2_kernel<true><<<dim3(DD / BN, EE * MBMAX), 256, 0, stream>>>(
        hex, exp_w2, exp_b2, counts, basep, ybuf);

    combine_kernel<<<TT, 256, 0, stream>>>(tslot, twt, basep, ybuf, out);
}
